// NGPNeRFNetwork_78847009620519
// MI455X (gfx1250) — matrix operand order repaired and checked by reading, not compiled, not run on hardware
//
#include <hip/hip_runtime.h>
#include <math.h>

typedef _Float16 h16v __attribute__((ext_vector_type(16)));
typedef _Float16 h8v  __attribute__((ext_vector_type(8)));
typedef float    f8v  __attribute__((ext_vector_type(8)));

#define TBITS 19
#define TMASK ((1u << TBITS) - 1u)
#define HP1 2654435761u
#define HP2 805459861u
#define SCALE_C 0.5f

__device__ __forceinline__ f8v wmma_f16(h16v a, h16v b, f8v c) {
  return __builtin_amdgcn_wmma_f32_16x16x32_f16(false, a, false, b, (short)0, c, false, false);
}

// Keep WMMA batches together: nothing may be scheduled across this point.
__device__ __forceinline__ void sched_fence() {
#if defined(__has_builtin)
#if __has_builtin(__builtin_amdgcn_sched_barrier)
  __builtin_amdgcn_sched_barrier(0);
#endif
#endif
}

// A fragment from a MxK (stride = K halves) row-major LDS activation buffer.
// ISA 16-bit A layout: lane<16 -> K = {0..7, 16..23}+kc*32 ; lane>=16 -> K = {8..15, 24..31}+kc*32
__device__ __forceinline__ h16v load_a(const _Float16* buf, int strideK, int lane, int kc) {
  const int m = lane & 15;
  const int koff = kc * 32 + ((lane < 16) ? 0 : 8);
  const _Float16* p = buf + m * strideK + koff;
  union { h16v v; h8v h[2]; } u;
  u.h[0] = *(const h8v*)(p);
  u.h[1] = *(const h8v*)(p + 16);
  return u.v;
}

// B fragment from transposed weights W^T [Nout][Kin] in LDS.
// ISA 16-bit B layout: lane<16 -> K = {0..7, 16..23}+kc*32 ; lane>=16 -> K = {8..15, 24..31}+kc*32 ; N = lane%16
__device__ __forceinline__ h16v load_b(const _Float16* wT, int Kin, int lane, int nc, int kc) {
  const int nn = nc * 16 + (lane & 15);
  const int kb = kc * 32 + ((lane < 16) ? 0 : 8);
  const _Float16* q = wT + nn * Kin + kb;
  union { h16v v; h8v h[2]; } u;
  u.h[0] = *(const h8v*)(q);
  u.h[1] = *(const h8v*)(q + 16);
  return u.v;
}

__device__ __forceinline__ unsigned pack2h(float a, float b) {
  union { _Float16 h[2]; unsigned u; } p;
  p.h[0] = (_Float16)a;
  p.h[1] = (_Float16)b;
  return p.u;
}

// bias + relu + f16 convert + scatter to LDS activation buffer (both 16-row tiles)
__device__ __forceinline__ void store_act2(_Float16* dst, int strideN, int mb, int col, int nc,
                                           f8v c0, f8v c1, float bias) {
#pragma unroll
  for (int r = 0; r < 8; ++r) {
    float v0 = c0[r] + bias; v0 = v0 > 0.0f ? v0 : 0.0f;
    float v1 = c1[r] + bias; v1 = v1 > 0.0f ? v1 : 0.0f;
    dst[(mb + r) * strideN + nc * 16 + col] = (_Float16)v0;
    dst[(16 + mb + r) * strideN + nc * 16 + col] = (_Float16)v1;
  }
}

// per-level hash-grid setup: compute 8 corner hashes and ISSUE the 8 gathers
struct LvlCtx {
  float rx, ry, rz;
  float2 fe[8];
};

__device__ __forceinline__ void prep_level(const float* __restrict__ table, float res, int lvl,
                                           float x0, float y0, float z0, LvlCtx& c) {
  const float px = x0 * res, py = y0 * res, pz = z0 * res;
  const float fx = floorf(px), fy = floorf(py), fz = floorf(pz);
  c.rx = px - fx; c.ry = py - fy; c.rz = pz - fz;
  const unsigned cx = (unsigned)(int)fx, cy = (unsigned)(int)fy, cz = (unsigned)(int)fz;
  const float* tb = table + ((size_t)lvl << (TBITS + 1));
  unsigned hsh[8];
#pragma unroll
  for (int cnr = 0; cnr < 8; ++cnr) {
    const unsigned ox = (cnr >> 2) & 1u, oy = (cnr >> 1) & 1u, oz = cnr & 1u;
    hsh[cnr] = ((cx + ox) ^ ((cy + oy) * HP1) ^ ((cz + oz) * HP2)) & TMASK;
  }
#pragma unroll
  for (int cnr = 0; cnr < 8; ++cnr)
    c.fe[cnr] = *(const float2*)(tb + ((size_t)hsh[cnr] << 1));
}

__global__ __launch_bounds__(128) void ngp_fused_kernel(
    const float* __restrict__ xin, const float* __restrict__ din,
    const float* __restrict__ table,
    const float* __restrict__ dW0, const float* __restrict__ db0,
    const float* __restrict__ dW1, const float* __restrict__ db1,
    const float* __restrict__ cW0, const float* __restrict__ cb0,
    const float* __restrict__ cW1, const float* __restrict__ cb1,
    const float* __restrict__ cW2, const float* __restrict__ cb2,
    float* __restrict__ out, int n)
{
  // ---- LDS: f16 transposed weights (shared by block) + per-wave activation buffers
  __shared__ __align__(16) _Float16 sWd0[64 * 32];
  __shared__ __align__(16) _Float16 sWd1[16 * 64];
  __shared__ __align__(16) _Float16 sWc0[64 * 32];
  __shared__ __align__(16) _Float16 sWc1[64 * 64];
  __shared__ __align__(16) _Float16 sWc2[16 * 64];
  __shared__ float sBd0[64], sBd1[16], sBc0[64], sBc1[64], sBc2[4];
  __shared__ __align__(16) _Float16 sBufA[4][32 * 32];  // enc, later hh = [sh | h] (two tiles)
  __shared__ __align__(16) _Float16 sBufH[4][32 * 64];  // hidden 64-wide
  __shared__ __align__(16) _Float16 sBufG[4][32 * 64];  // hidden 64-wide (2nd)

  const int tid = threadIdx.x;

  // ---- one-time: load + transpose + f16-convert weights into LDS
  for (int i = tid; i < 64 * 32; i += 128) { int nn = i >> 5, kk = i & 31; sWd0[i] = (_Float16)dW0[kk * 64 + nn]; }
  for (int i = tid; i < 16 * 64; i += 128) { int nn = i >> 6, kk = i & 63; sWd1[i] = (_Float16)dW1[kk * 16 + nn]; }
  for (int i = tid; i < 64 * 32; i += 128) { int nn = i >> 5, kk = i & 31; sWc0[i] = (_Float16)cW0[kk * 64 + nn]; }
  for (int i = tid; i < 64 * 64; i += 128) { int nn = i >> 6, kk = i & 63; sWc1[i] = (_Float16)cW1[kk * 64 + nn]; }
  for (int i = tid; i < 16 * 64; i += 128) {
    int nn = i >> 6, kk = i & 63;
    sWc2[i] = (nn < 3) ? (_Float16)cW2[kk * 3 + nn] : (_Float16)0.0f;
  }
  if (tid < 64) sBd0[tid] = db0[tid];
  if (tid < 16) sBd1[tid] = db1[tid];
  if (tid < 64) sBc0[tid] = cb0[tid];
  if (tid < 64) sBc1[tid] = cb1[tid];
  if (tid < 4)  sBc2[tid] = (tid < 3) ? cb2[tid] : 0.0f;
  __syncthreads();

  const int wave = tid >> 5;
  const int lane = tid & 31;
  const int col = lane & 15;            // N index inside a 16-col chunk
  const int mb = (lane < 16) ? 0 : 8;   // C/D fragment row base
  _Float16* bufA = sBufA[wave];
  _Float16* bufH = sBufH[wave];
  _Float16* bufG = sBufG[wave];
  unsigned* bufA32 = (unsigned*)bufA;

  // RES[l] = floor(16 * 128^(l/15))
  const float RESF[16] = {16.0f, 22.0f, 30.0f, 42.0f, 58.0f, 80.0f, 111.0f, 153.0f,
                          212.0f, 294.0f, 406.0f, 561.0f, 776.0f, 1072.0f, 1482.0f, 2048.0f};

  const int ngroups = n >> 7;  // 128 points per block-iteration (4 waves x 32)
  for (int g = blockIdx.x; g < ngroups; g += gridDim.x) {
    const int base = (g * 4 + wave) * 32;
    const int p = lane;       // point owned by this lane for hash/SH (0..31)
    const int gp = base + p;

    // ================= hash-grid encoding: 1 lane per point, 16 levels, SW-pipelined ===
    {
      const float x0 = (xin[gp * 3 + 0] + SCALE_C) * (0.5f / SCALE_C);
      const float y0 = (xin[gp * 3 + 1] + SCALE_C) * (0.5f / SCALE_C);
      const float z0 = (xin[gp * 3 + 2] + SCALE_C) * (0.5f / SCALE_C);
      LvlCtx cur, nxt;
      prep_level(table, RESF[0], 0, x0, y0, z0, cur);
#pragma unroll
      for (int lvl = 0; lvl < 16; ++lvl) {
        if (lvl < 15) prep_level(table, RESF[lvl + 1], lvl + 1, x0, y0, z0, nxt);
        float f0 = 0.0f, f1 = 0.0f;
#pragma unroll
        for (int cnr = 0; cnr < 8; ++cnr) {
          const unsigned ox = (cnr >> 2) & 1u, oy = (cnr >> 1) & 1u, oz = cnr & 1u;
          const float w = (ox ? cur.rx : 1.0f - cur.rx) *
                          (oy ? cur.ry : 1.0f - cur.ry) *
                          (oz ? cur.rz : 1.0f - cur.rz);
          f0 = fmaf(w, cur.fe[cnr].x, f0);
          f1 = fmaf(w, cur.fe[cnr].y, f1);
        }
        bufA32[p * 16 + lvl] = pack2h(f0, f1);  // one dword per level
        cur = nxt;
      }
    }

    // ================= density L0: enc(32x32) @ dW0 -> relu -> bufH(32x64) =============
    {
      const h16v a0 = load_a(bufA, 32, lane, 0);
      const h16v a1 = load_a(bufA + 16 * 32, 32, lane, 0);
#pragma unroll
      for (int nc = 0; nc < 4; nc += 2) {
        const h16v bA = load_b(sWd0, 32, lane, nc, 0);
        const h16v bB = load_b(sWd0, 32, lane, nc + 1, 0);
        f8v c0a = {}, c1a = {}, c0b = {}, c1b = {};
        c0a = wmma_f16(a0, bA, c0a);
        c1a = wmma_f16(a1, bA, c1a);
        c0b = wmma_f16(a0, bB, c0b);
        c1b = wmma_f16(a1, bB, c1b);
        sched_fence();
        store_act2(bufH, 64, mb, col, nc,     c0a, c1a, sBd0[nc * 16 + col]);
        store_act2(bufH, 64, mb, col, nc + 1, c0b, c1b, sBd0[(nc + 1) * 16 + col]);
      }
    }

    // ================= density L1: bufH(32x64) @ dW1 -> h(32x16); sigma = exp(h[:,0]) ==
    {
      const h16v a00 = load_a(bufH, 64, lane, 0);
      const h16v a01 = load_a(bufH, 64, lane, 1);
      const h16v a10 = load_a(bufH + 16 * 64, 64, lane, 0);
      const h16v a11 = load_a(bufH + 16 * 64, 64, lane, 1);
      const h16v b0 = load_b(sWd1, 64, lane, 0, 0);
      const h16v b1 = load_b(sWd1, 64, lane, 0, 1);
      f8v c0 = {}, c1 = {};
      c0 = wmma_f16(a00, b0, c0);
      c1 = wmma_f16(a10, b0, c1);
      c0 = wmma_f16(a01, b1, c0);
      c1 = wmma_f16(a11, b1, c1);
      sched_fence();
      const float bias = sBd1[col];
      if (col == 0) {
#pragma unroll
        for (int r = 0; r < 8; ++r) {
          out[base + mb + r] = expf(c0[r] + bias);
          out[base + 16 + mb + r] = expf(c1[r] + bias);
        }
      }
      // h goes to columns 16..31 of hh (overwrites consumed enc)
#pragma unroll
      for (int r = 0; r < 8; ++r) {
        bufA[(mb + r) * 32 + 16 + col] = (_Float16)(c0[r] + bias);
        bufA[(16 + mb + r) * 32 + 16 + col] = (_Float16)(c1[r] + bias);
      }
    }

    // ================= SH deg4: columns 0..15 of hh (1 lane per point) =================
    {
      const float dx = din[gp * 3 + 0], dy = din[gp * 3 + 1], dz = din[gp * 3 + 2];
      const float inv = rsqrtf(dx * dx + dy * dy + dz * dz);
      const float sx = dx * inv, sy = dy * inv, sz = dz * inv;
      const float x2 = sx * sx, y2 = sy * sy, z2 = sz * sz;
      const float xy = sx * sy, yz = sy * sz, xz = sx * sz;
      union { _Float16 h[8]; h8v v; } lo, hi;
      lo.h[0] = (_Float16)(0.28209479177387814f);
      lo.h[1] = (_Float16)(-0.48860251190291987f * sy);
      lo.h[2] = (_Float16)( 0.48860251190291987f * sz);
      lo.h[3] = (_Float16)(-0.48860251190291987f * sx);
      lo.h[4] = (_Float16)( 1.0925484305920792f * xy);
      lo.h[5] = (_Float16)(-1.0925484305920792f * yz);
      lo.h[6] = (_Float16)( 0.94617469575756f * z2 - 0.31539156525252f);
      lo.h[7] = (_Float16)(-1.0925484305920792f * xz);
      hi.h[0] = (_Float16)(0.5462742152960396f * (x2 - y2));
      hi.h[1] = (_Float16)(0.5900435899266435f * sy * (-3.0f * x2 + y2));
      hi.h[2] = (_Float16)(2.890611442640554f * xy * sz);
      hi.h[3] = (_Float16)(0.4570457994644657f * sy * (1.0f - 5.0f * z2));
      hi.h[4] = (_Float16)(0.3731763325901154f * sz * (5.0f * z2 - 3.0f));
      hi.h[5] = (_Float16)(0.4570457994644657f * sx * (1.0f - 5.0f * z2));
      hi.h[6] = (_Float16)(1.445305721320277f * sz * (x2 - y2));
      hi.h[7] = (_Float16)(0.5900435899266435f * sx * (-x2 + 3.0f * y2));
      *(h8v*)&bufA[p * 32 + 0] = lo.v;
      *(h8v*)&bufA[p * 32 + 8] = hi.v;
    }

    // ================= color L0: hh(32x32) @ cW0 -> relu -> bufH(32x64) ================
    {
      const h16v a0 = load_a(bufA, 32, lane, 0);
      const h16v a1 = load_a(bufA + 16 * 32, 32, lane, 0);
#pragma unroll
      for (int nc = 0; nc < 4; nc += 2) {
        const h16v bA = load_b(sWc0, 32, lane, nc, 0);
        const h16v bB = load_b(sWc0, 32, lane, nc + 1, 0);
        f8v c0a = {}, c1a = {}, c0b = {}, c1b = {};
        c0a = wmma_f16(a0, bA, c0a);
        c1a = wmma_f16(a1, bA, c1a);
        c0b = wmma_f16(a0, bB, c0b);
        c1b = wmma_f16(a1, bB, c1b);
        sched_fence();
        store_act2(bufH, 64, mb, col, nc,     c0a, c1a, sBc0[nc * 16 + col]);
        store_act2(bufH, 64, mb, col, nc + 1, c0b, c1b, sBc0[(nc + 1) * 16 + col]);
      }
    }

    // ================= color L1: bufH(32x64) @ cW1 -> relu -> bufG(32x64) ==============
    {
      const h16v a00 = load_a(bufH, 64, lane, 0);
      const h16v a01 = load_a(bufH, 64, lane, 1);
      const h16v a10 = load_a(bufH + 16 * 64, 64, lane, 0);
      const h16v a11 = load_a(bufH + 16 * 64, 64, lane, 1);
#pragma unroll
      for (int nc = 0; nc < 4; nc += 2) {
        const h16v b0A = load_b(sWc1, 64, lane, nc, 0);
        const h16v b1A = load_b(sWc1, 64, lane, nc, 1);
        const h16v b0B = load_b(sWc1, 64, lane, nc + 1, 0);
        const h16v b1B = load_b(sWc1, 64, lane, nc + 1, 1);
        f8v c0a = {}, c1a = {}, c0b = {}, c1b = {};
        c0a = wmma_f16(a00, b0A, c0a);
        c1a = wmma_f16(a10, b0A, c1a);
        c0b = wmma_f16(a00, b0B, c0b);
        c1b = wmma_f16(a10, b0B, c1b);
        c0a = wmma_f16(a01, b1A, c0a);
        c1a = wmma_f16(a11, b1A, c1a);
        c0b = wmma_f16(a01, b1B, c0b);
        c1b = wmma_f16(a11, b1B, c1b);
        sched_fence();
        store_act2(bufG, 64, mb, col, nc,     c0a, c1a, sBc1[nc * 16 + col]);
        store_act2(bufG, 64, mb, col, nc + 1, c0b, c1b, sBc1[(nc + 1) * 16 + col]);
      }
    }

    // ================= color L2: bufG(32x64) @ cW2(pad 16 cols) -> sigmoid -> out ======
    {
      const h16v a00 = load_a(bufG, 64, lane, 0);
      const h16v a01 = load_a(bufG, 64, lane, 1);
      const h16v a10 = load_a(bufG + 16 * 64, 64, lane, 0);
      const h16v a11 = load_a(bufG + 16 * 64, 64, lane, 1);
      const h16v b0 = load_b(sWc2, 64, lane, 0, 0);
      const h16v b1 = load_b(sWc2, 64, lane, 0, 1);
      f8v c0 = {}, c1 = {};
      c0 = wmma_f16(a00, b0, c0);
      c1 = wmma_f16(a10, b0, c1);
      c0 = wmma_f16(a01, b1, c0);
      c1 = wmma_f16(a11, b1, c1);
      sched_fence();
      if (col < 3) {
        const float bias = sBc2[col];
#pragma unroll
        for (int r = 0; r < 8; ++r) {
          const float v0 = c0[r] + bias;
          const float v1 = c1[r] + bias;
          out[(size_t)n + (size_t)(base + mb + r) * 3 + col] = 1.0f / (1.0f + expf(-v0));
          out[(size_t)n + (size_t)(base + 16 + mb + r) * 3 + col] = 1.0f / (1.0f + expf(-v1));
        }
      }
    }
  }
}

extern "C" void kernel_launch(void* const* d_in, const int* in_sizes, int n_in,
                              void* d_out, int out_size, void* d_ws, size_t ws_size,
                              hipStream_t stream) {
  const float* x     = (const float*)d_in[0];
  const float* d     = (const float*)d_in[1];
  const float* table = (const float*)d_in[2];
  const float* dW0   = (const float*)d_in[3];
  const float* db0   = (const float*)d_in[4];
  const float* dW1   = (const float*)d_in[5];
  const float* db1   = (const float*)d_in[6];
  const float* cW0   = (const float*)d_in[7];
  const float* cb0   = (const float*)d_in[8];
  const float* cW1   = (const float*)d_in[9];
  const float* cb1   = (const float*)d_in[10];
  const float* cW2   = (const float*)d_in[11];
  const float* cb2   = (const float*)d_in[12];
  float* out = (float*)d_out;
  const int n = in_sizes[0] / 3;

  // Persistent blocks: amortize the weight-transpose-to-LDS over many 128-point groups.
  const int grid = 2048;
  ngp_fused_kernel<<<grid, 128, 0, stream>>>(x, d, table, dW0, db0, dW1, db1,
                                             cW0, cb0, cW1, cb1, cW2, cb2, out, n);
}
